// AllNodes_55843164783208
// MI455X (gfx1250) — matrix operand order repaired and checked by reading, not compiled, not run on hardware
//
#include <hip/hip_runtime.h>

typedef __bf16 bf16;
typedef __attribute__((ext_vector_type(16))) __bf16 v16bf;
typedef __attribute__((ext_vector_type(8)))  float  v8f;

#define DIM   128
#define PITCH 130          // bf16 elements per LDS row: 65 dwords -> conflict-free

// ---------------------------------------------------------------------------
// Kernel 1: copy node_tensor -> out (float4, fully coalesced, HBM-bound)
// ---------------------------------------------------------------------------
__global__ void copy_nodes_kernel(const float4* __restrict__ src,
                                  float4* __restrict__ dst, int n4) {
    int i = blockIdx.x * blockDim.x + threadIdx.x;
    if (i < n4) dst[i] = src[i];
}

// ---------------------------------------------------------------------------
// Kernel 2: gather -> MLP (WMMA bf16, f32 acc) -> scatter into out
// One wave (32 lanes) owns a 32-row slab = two 16-row A-tiles, so every
// B-fragment loaded from LDS feeds TWO v_wmma issues (halves LDS B traffic).
// ---------------------------------------------------------------------------
__global__ __launch_bounds__(256)
void mlp_scatter_kernel(const float* __restrict__ node,
                        const float* __restrict__ W1,
                        const float* __restrict__ b1,
                        const float* __restrict__ W2,
                        const float* __restrict__ b2,
                        const int*   __restrict__ part,
                        float* __restrict__ out,
                        int P) {
    __shared__ bf16 Wt1[DIM * PITCH];           // W1 transposed: Wt1[n*PITCH + k]
    __shared__ bf16 Wt2[DIM * PITCH];           // W2 transposed
    __shared__ bf16 Hst[8][32 * PITCH];         // per-wave h staging: [m][n], 32 rows

    const int tid  = threadIdx.x;
    const int wave = tid >> 5;
    const int lane = tid & 31;
    const int half = lane >> 4;                 // 0: lanes 0-15, 1: lanes 16-31
    const int mrow = lane & 15;                 // row within a 16-row tile

    const int slab = blockIdx.x * 8 + wave;     // 32-row slab id

    // ---- gather: row indices for this lane's A-rows (clamped; duplicated
    // rows produce bit-identical results, so overflow work is harmless) ------
    int j0 = slab * 32 + mrow;
    int j1 = j0 + 16;
    if (j0 > P - 1) j0 = P - 1;
    if (j1 > P - 1) j1 = P - 1;
    const int row0 = part[j0];
    const int row1 = part[j1];
    const float* xrow0 = node + (size_t)row0 * DIM;
    const float* xrow1 = node + (size_t)row1 * DIM;

    // A fragments: 4 chunks of K=32 each, ISA 16-bit A 16x32 layout.
    // element pair p -> K = (p<4 ? 2p : 2p+8) + 8*half + 32*chunk
    v16bf a0[4], a1[4];
#pragma unroll
    for (int c = 0; c < 4; ++c) {
#pragma unroll
        for (int p = 0; p < 8; ++p) {
            const int kp = ((p < 4) ? 2 * p : 2 * p + 8) + half * 8 + 32 * c;
            const float2 v0 = *(const float2*)(xrow0 + kp);
            const float2 v1 = *(const float2*)(xrow1 + kp);
            a0[c][2 * p]     = (bf16)v0.x;
            a0[c][2 * p + 1] = (bf16)v0.y;
            a1[c][2 * p]     = (bf16)v1.x;
            a1[c][2 * p + 1] = (bf16)v1.y;
        }
    }

    // Broadcast all 16 row indices of each tile-half for the final scatter.
    // C/D layout: VGPR r holds M = r + 8*half.
    int rowv0[8], rowv1[8];
#pragma unroll
    for (int r = 0; r < 8; ++r) {
        rowv0[r] = __builtin_amdgcn_ds_bpermute((r + half * 8) * 4, row0);
        rowv1[r] = __builtin_amdgcn_ds_bpermute((r + half * 8) * 4, row1);
    }

    // ---- stage both weight matrices to LDS, transposed, as bf16 ------------
    for (int i = tid; i < (DIM * DIM) / 4; i += 256) {
        const int k  = i >> 5;                  // source row (K index)
        const int n4 = (i & 31) << 2;           // source col base (N index)
        const float4 w1v = *(const float4*)(W1 + k * DIM + n4);
        const float4 w2v = *(const float4*)(W2 + k * DIM + n4);
        Wt1[(n4 + 0) * PITCH + k] = (bf16)w1v.x;
        Wt1[(n4 + 1) * PITCH + k] = (bf16)w1v.y;
        Wt1[(n4 + 2) * PITCH + k] = (bf16)w1v.z;
        Wt1[(n4 + 3) * PITCH + k] = (bf16)w1v.w;
        Wt2[(n4 + 0) * PITCH + k] = (bf16)w2v.x;
        Wt2[(n4 + 1) * PITCH + k] = (bf16)w2v.y;
        Wt2[(n4 + 2) * PITCH + k] = (bf16)w2v.z;
        Wt2[(n4 + 3) * PITCH + k] = (bf16)w2v.w;
    }
    __syncthreads();

    bf16* hst = Hst[wave];

    // ---- layer 1: h = relu(x @ W1 + b1), staged to LDS as bf16 -------------
#pragma unroll 1
    for (int t = 0; t < 8; ++t) {               // 8 output column tiles of 16
        const float bias = b1[t * 16 + mrow];
        v8f acc0 = {bias, bias, bias, bias, bias, bias, bias, bias};
        v8f acc1 = acc0;
#pragma unroll
        for (int c = 0; c < 4; ++c) {
            // B fragment (KxN): element pair p -> K = (p<4 ? 2p : 2p+8) + 8*half (+32*c), N = mrow+16t
            v16bf bf;
            const int nb = (t * 16 + mrow) * PITCH;
#pragma unroll
            for (int p = 0; p < 8; ++p) {
                const int off = nb + ((p < 4) ? 2 * p : 2 * p + 8) + half * 8 + 32 * c;
                bf[2 * p]     = Wt1[off];
                bf[2 * p + 1] = Wt1[off + 1];
            }
            acc0 = __builtin_amdgcn_wmma_f32_16x16x32_bf16(
                false, a0[c], false, bf, (short)0, acc0, false, false);
            acc1 = __builtin_amdgcn_wmma_f32_16x16x32_bf16(
                false, a1[c], false, bf, (short)0, acc1, false, false);
        }
#pragma unroll
        for (int r = 0; r < 8; ++r) {
            hst[(r + half * 8) * PITCH + t * 16 + mrow]      = (bf16)fmaxf(acc0[r], 0.f);
            hst[(r + half * 8 + 16) * PITCH + t * 16 + mrow] = (bf16)fmaxf(acc1[r], 0.f);
        }
    }

    // same-wave LDS ops are in-order (DScnt); fence the compiler + counters
    asm volatile("s_wait_dscnt 0" ::: "memory");
    __builtin_amdgcn_wave_barrier();

    // ---- rebuild A fragments from staged h ---------------------------------
#pragma unroll
    for (int c = 0; c < 4; ++c) {
#pragma unroll
        for (int p = 0; p < 8; ++p) {
            const int kp   = ((p < 4) ? 2 * p : 2 * p + 8) + half * 8 + 32 * c;
            const int off0 = mrow * PITCH + kp;
            const int off1 = (mrow + 16) * PITCH + kp;
            a0[c][2 * p]     = hst[off0];
            a0[c][2 * p + 1] = hst[off0 + 1];
            a1[c][2 * p]     = hst[off1];
            a1[c][2 * p + 1] = hst[off1 + 1];
        }
    }

    // ---- layer 2: y = h @ W2 + b2, scatter f32 rows into out ---------------
#pragma unroll 1
    for (int t = 0; t < 8; ++t) {
        const float bias = b2[t * 16 + mrow];
        v8f acc0 = {bias, bias, bias, bias, bias, bias, bias, bias};
        v8f acc1 = acc0;
#pragma unroll
        for (int c = 0; c < 4; ++c) {
            v16bf bf;
            const int nb = (t * 16 + mrow) * PITCH;
#pragma unroll
            for (int p = 0; p < 8; ++p) {
                const int off = nb + ((p < 4) ? 2 * p : 2 * p + 8) + half * 8 + 32 * c;
                bf[2 * p]     = Wt2[off];
                bf[2 * p + 1] = Wt2[off + 1];
            }
            acc0 = __builtin_amdgcn_wmma_f32_16x16x32_bf16(
                false, a0[c], false, bf, (short)0, acc0, false, false);
            acc1 = __builtin_amdgcn_wmma_f32_16x16x32_bf16(
                false, a1[c], false, bf, (short)0, acc1, false, false);
        }
#pragma unroll
        for (int r = 0; r < 8; ++r) {
            out[(size_t)rowv0[r] * DIM + t * 16 + mrow] = acc0[r];
            out[(size_t)rowv1[r] * DIM + t * 16 + mrow] = acc1[r];
        }
    }
}

// ---------------------------------------------------------------------------
extern "C" void kernel_launch(void* const* d_in, const int* in_sizes, int n_in,
                              void* d_out, int out_size, void* d_ws, size_t ws_size,
                              hipStream_t stream) {
    const float* node = (const float*)d_in[0];
    const float* W1   = (const float*)d_in[1];
    const float* b1   = (const float*)d_in[2];
    const float* W2   = (const float*)d_in[3];
    const float* b2   = (const float*)d_in[4];
    const int*   part = (const int*)d_in[5];
    float*       out  = (float*)d_out;

    // 1) full copy node_tensor -> out
    const int n4 = out_size / 4;                       // N*D/4 float4s
    copy_nodes_kernel<<<(n4 + 255) / 256, 256, 0, stream>>>(
        (const float4*)node, (float4*)out, n4);

    // 2) gather -> MLP -> scatter (one wave per 32-row slab)
    const int P     = in_sizes[5];
    const int slabs = (P + 31) / 32;
    const int blocks = (slabs + 7) / 8;                // 8 waves per block
    mlp_scatter_kernel<<<blocks, 256, 0, stream>>>(node, W1, b1, W2, b2,
                                                   part, out, P);
}
